// VectorQuantizer1D_27857157881909
// MI455X (gfx1250) — matrix operand order repaired and checked by reading, not compiled, not run on hardware
//
#include <hip/hip_runtime.h>
#include <hip/hip_bf16.h>
#include <math.h>

// ---------------------------------------------------------------------------
// VQ (vector quantizer) for MI455X / gfx1250.
//  - argmin via bf16x2-split WMMA (3x v_wmma_f32_16x16x32_bf16 per 32-wide
//    K-step, ~fp32 accuracy)
//  - 112 tokens resident in LDS per WG (7 compute waves + 1 helper wave)
//  - codebook pre-split to bf16 hi/lo in global; 16-code W tiles streamed
//    through a DOUBLE-BUFFERED LDS stage via async global->LDS DMA
//    (ASYNCcnt), overlapped with WMMA compute.
// ---------------------------------------------------------------------------

typedef __bf16 bf16_t;
typedef __attribute__((ext_vector_type(16))) __bf16 v16bf;
typedef __attribute__((ext_vector_type(8)))  __bf16 v8bf;
typedef __attribute__((ext_vector_type(8)))  float  v8f;
typedef int v4i __attribute__((__vector_size__(4 * sizeof(int))));

#define NUM_EMB    4096
#define DIM        512
#define NTOK       32768
#define ROWPAD     520          // 512 + 8 halfs -> +4 dwords/row, conflict-free ds_load_b128
#define TOK_PER_WG 112          // 7 waves x 16-token row tile (+1 stager wave)
#define NTILES     (NUM_EMB / 16)
#define COMMIT     0.25f

#define GAS __attribute__((address_space(1)))
#define LAS __attribute__((address_space(3)))

#if defined(__HIP_DEVICE_COMPILE__) && __has_builtin(__builtin_amdgcn_global_load_async_to_lds_b128)
#define ASYNC_LDS 1
#endif

__device__ inline void cp16_to_lds(void* l, const void* g) {
#ifdef ASYNC_LDS
    __builtin_amdgcn_global_load_async_to_lds_b128((GAS v4i*)g, (LAS v4i*)l, 0, 0);
#else
    *(uint4*)l = *(const uint4*)g;
#endif
}

__device__ inline void wait_async_copies() {
#ifdef ASYNC_LDS
#if __has_builtin(__builtin_amdgcn_s_wait_asynccnt)
    __builtin_amdgcn_s_wait_asynccnt(0);
#else
    asm volatile("s_wait_asynccnt 0x0" ::: "memory");
#endif
#endif
}

// ---------------- codebook pre-split: f32 -> bf16 hi + bf16 lo -------------
__global__ __launch_bounds__(256) void wsplit_kernel(const float* __restrict__ W,
                                                     bf16_t* __restrict__ WHg,
                                                     bf16_t* __restrict__ WLg) {
    int i = blockIdx.x * 256 + threadIdx.x;          // over float4 elements
    float4 v = ((const float4*)W)[i];
    float vals[4] = {v.x, v.y, v.z, v.w};
    #pragma unroll
    for (int j = 0; j < 4; ++j) {
        float f = vals[j];
        bf16_t h = (bf16_t)f;
        WHg[i * 4 + j] = h;
        WLg[i * 4 + j] = (bf16_t)(f - (float)h);
    }
}

// --------------------------- codebook row norms ----------------------------
__global__ __launch_bounds__(256) void wnorm_kernel(const float* __restrict__ W,
                                                    float* __restrict__ wnorm) {
    int wv = threadIdx.x >> 5, lane = threadIdx.x & 31;
    int row = blockIdx.x * 8 + wv;
    const float* p = W + (size_t)row * DIM;
    float s = 0.f;
    for (int j = lane; j < DIM; j += 32) { float v = p[j]; s += v * v; }
    for (int m = 16; m > 0; m >>= 1) s += __shfl_xor(s, m, 32);
    if (lane == 0) wnorm[row] = s;
}

// ------------------------------ zero counts --------------------------------
__global__ __launch_bounds__(256) void zero_counts_kernel(unsigned* __restrict__ counts) {
    for (int c = blockIdx.x * 256 + threadIdx.x; c < NUM_EMB; c += gridDim.x * 256)
        counts[c] = 0u;
}

// ------------------------- main argmin GEMM kernel -------------------------
// score(token, code) = ||w||^2 - 2 * dot(x, w); argmin over codes.
__global__ __launch_bounds__(256) void vq_argmin_kernel(const float* __restrict__ X,
                                                        const bf16_t* __restrict__ WHg,
                                                        const bf16_t* __restrict__ WLg,
                                                        const float* __restrict__ wnorm,
                                                        int* __restrict__ out_idx) {
    extern __shared__ char smem[];
    bf16_t* XH  = (bf16_t*)smem;                    // [112][520] hi halves of x
    bf16_t* XL  = XH + TOK_PER_WG * ROWPAD;         // [112][520] lo halves of x
    bf16_t* WB0 = XL + TOK_PER_WG * ROWPAD;         // W tile buf 0: hi[16][520] lo[16][520]
    bf16_t* WB1 = WB0 + 2 * 16 * ROWPAD;            // W tile buf 1

    const int tid     = threadIdx.x;
    const int lane    = tid & 31;
    const int wv      = tid >> 5;
    const int tokBase = blockIdx.x * TOK_PER_WG;
    const int nrows   = min(TOK_PER_WG, NTOK - tokBase);
    const bool active = (wv < 7) && (tokBase + wv * 16 < NTOK);

    // ---- stage the X block into LDS as bf16 hi/lo split (once per WG) ----
    {
        const float4* X4 = (const float4*)(X + (size_t)tokBase * DIM);
        for (int i = tid; i < nrows * (DIM / 4); i += 256) {
            int row = i >> 7;               // 128 float4 per row
            int c4  = (i & 127) << 2;
            float4 v = X4[i];
            float vals[4] = {v.x, v.y, v.z, v.w};
            #pragma unroll
            for (int j = 0; j < 4; ++j) {
                float f = vals[j];
                bf16_t h = (bf16_t)f;
                XH[row * ROWPAD + c4 + j] = h;
                XL[row * ROWPAD + c4 + j] = (bf16_t)(f - (float)h);
            }
        }
    }

    // ---- W tile stager: 16 codes x 512 halfs, hi then lo, padded rows ----
    // 2048 x b128 chunks per tile (8 per thread) via async DMA to LDS.
    auto stage_w = [&](bf16_t* wbuf, int t) {
        #pragma unroll
        for (int k = 0; k < 8; ++k) {
            int c   = tid + k * 256;        // 0..2047
            int arr = c >> 10;              // 0 = hi, 1 = lo
            int r   = (c & 1023) >> 6;      // row (code) 0..15
            int col = (c & 63) << 3;        // half offset, step 8 (16B)
            const bf16_t* g = (arr ? WLg : WHg) + ((size_t)(t * 16 + r) * DIM + col);
            bf16_t*       l = wbuf + arr * (16 * ROWPAD) + r * ROWPAD + col;
            cp16_to_lds(l, g);
        }
    };

    stage_w(WB0, 0);
    wait_async_copies();
    __syncthreads();

    float bestv[8];
    int   besti[8];
    #pragma unroll
    for (int r = 0; r < 8; ++r) { bestv[r] = __builtin_huge_valf(); besti[r] = 0; }

    // fragment addressing (per ISA layouts, wave32):
    //  A 16x32 bf16: lanes<16 rows, K runs [d+0,8) & [d+16,8); lanes>=16: +8
    //  B 32x16 bf16: lanes<16 cols, K runs [d+0,8) & [d+16,8); lanes>=16: +8
    const int arow  = (lane & 15) * ROWPAD;
    const int akoff = (lane >> 4) << 3;     // 0 or 8
    const int bkoff = (lane >> 4) << 4;     // 0 or 16
    const bf16_t* aBaseH = XH + (wv * 16) * ROWPAD;
    const bf16_t* aBaseL = XL + (wv * 16) * ROWPAD;

    for (int t = 0; t < NTILES; ++t) {
        bf16_t* cur = (t & 1) ? WB1 : WB0;
        bf16_t* nxt = (t & 1) ? WB0 : WB1;
        if (t + 1 < NTILES) stage_w(nxt, t + 1);   // DMA next tile, overlapped

        if (active) {
            const bf16_t* whBase = cur;
            const bf16_t* wlBase = cur + 16 * ROWPAD;
            float wn = wnorm[t * 16 + (lane & 15)];

            v8f c = {0.f, 0.f, 0.f, 0.f, 0.f, 0.f, 0.f, 0.f};
            #pragma unroll 4
            for (int d = 0; d < DIM; d += 32) {
                v16bf ah, al, bh, bl;
                {   // A fragments (hi, lo)
                    v8bf x0 = *(const v8bf*)(aBaseH + arow + d + akoff);
                    v8bf x1 = *(const v8bf*)(aBaseH + arow + d + 16 + akoff);
                    v8bf y0 = *(const v8bf*)(aBaseL + arow + d + akoff);
                    v8bf y1 = *(const v8bf*)(aBaseL + arow + d + 16 + akoff);
                    ah = __builtin_shufflevector(x0, x1, 0,1,2,3,4,5,6,7,8,9,10,11,12,13,14,15);
                    al = __builtin_shufflevector(y0, y1, 0,1,2,3,4,5,6,7,8,9,10,11,12,13,14,15);
                }
                {   // B fragments (hi, lo)
                    v8bf x0 = *(const v8bf*)(whBase + arow + d + akoff);
                    v8bf x1 = *(const v8bf*)(whBase + arow + d + 16 + akoff);
                    v8bf y0 = *(const v8bf*)(wlBase + arow + d + akoff);
                    v8bf y1 = *(const v8bf*)(wlBase + arow + d + 16 + akoff);
                    bh = __builtin_shufflevector(x0, x1, 0,1,2,3,4,5,6,7,8,9,10,11,12,13,14,15);
                    bl = __builtin_shufflevector(y0, y1, 0,1,2,3,4,5,6,7,8,9,10,11,12,13,14,15);
                }
                // bf16x2 split product: xh*wh + xl*wh + xh*wl  (~fp32 accuracy)
                c = __builtin_amdgcn_wmma_f32_16x16x32_bf16(false, ah, false, bh, (short)0, c, false, false);
                c = __builtin_amdgcn_wmma_f32_16x16x32_bf16(false, al, false, bh, (short)0, c, false, false);
                c = __builtin_amdgcn_wmma_f32_16x16x32_bf16(false, ah, false, bl, (short)0, c, false, false);
            }

            int code = t * 16 + (lane & 15);
            #pragma unroll
            for (int r = 0; r < 8; ++r) {
                float s = wn - 2.0f * c[r];
                if (s < bestv[r]) { bestv[r] = s; besti[r] = code; }  // strict <: keep first
            }
        }

        wait_async_copies();   // next buffer landed
        __syncthreads();       // everyone done reading cur
    }

    // reduce across the 16 column lanes of each half-wave (argmin w/ index tiebreak)
    if (active) {
        #pragma unroll
        for (int m = 1; m < 16; m <<= 1) {
            #pragma unroll
            for (int r = 0; r < 8; ++r) {
                float ov = __shfl_xor(bestv[r], m, 32);
                int   oi = __shfl_xor(besti[r], m, 32);
                if (ov < bestv[r] || (ov == bestv[r] && oi < besti[r])) {
                    bestv[r] = ov; besti[r] = oi;
                }
            }
        }
        if ((lane & 15) == 0) {
            int rbase = tokBase + wv * 16 + (lane >> 4) * 8;   // C row mapping
            #pragma unroll
            for (int r = 0; r < 8; ++r) out_idx[rbase + r] = besti[r];
        }
    }
}

// ----------------- gather, straight-through out, loss, histogram -----------
__global__ __launch_bounds__(256) void vq_gather_kernel(const float* __restrict__ X,
                                                        const float* __restrict__ W,
                                                        const int* __restrict__ idx,
                                                        float* __restrict__ outq,
                                                        float* __restrict__ loss,
                                                        float* __restrict__ out_idx_f,
                                                        unsigned* __restrict__ counts) {
    int wv = threadIdx.x >> 5, lane = threadIdx.x & 31;
    int tok = blockIdx.x * 8 + wv;
    int k = idx[tok];
    const float* wp = W + (size_t)k * DIM;
    const float* xp = X + (size_t)tok * DIM;
    float*       qp = outq + (size_t)tok * DIM;
    float acc = 0.f;
    for (int j = lane; j < DIM; j += 32) {
        float q = wp[j], x = xp[j];
        qp[j] = q;                     // quantized_st == quantized numerically
        float d = q - x;
        acc += d * d;
    }
    for (int m = 16; m > 0; m >>= 1) acc += __shfl_xor(acc, m, 32);
    if (lane == 0) {
        loss[tok]      = (1.0f + COMMIT) * acc * (1.0f / (float)DIM);
        out_idx_f[tok] = (float)k;
        atomicAdd(&counts[k], 1u);
    }
}

// ------------------------------- perplexity --------------------------------
__global__ __launch_bounds__(256) void perplexity_kernel(const unsigned* __restrict__ counts,
                                                         float* __restrict__ out) {
    __shared__ float red[256];
    float s = 0.f;
    for (int c = threadIdx.x; c < NUM_EMB; c += 256) {
        float p = (float)counts[c] * (1.0f / (float)NTOK);
        s += p * logf(p + 1e-10f);
    }
    red[threadIdx.x] = s;
    __syncthreads();
    for (int off = 128; off > 0; off >>= 1) {
        if (threadIdx.x < off) red[threadIdx.x] += red[threadIdx.x + off];
        __syncthreads();
    }
    if (threadIdx.x == 0) *out = expf(-red[0]);
}

// --------------------------------- launch ----------------------------------
extern "C" void kernel_launch(void* const* d_in, const int* in_sizes, int n_in,
                              void* d_out, int out_size, void* d_ws, size_t ws_size,
                              hipStream_t stream) {
    const float* X = (const float*)d_in[0];     // [NTOK, DIM]
    const float* W = (const float*)d_in[1];     // [NUM_EMB, DIM]

    float* out   = (float*)d_out;
    float* outq  = out;                                   // [NTOK*DIM]
    float* loss  = out + (size_t)NTOK * DIM;              // [NTOK]
    float* perp  = loss + NTOK;                           // [1]
    float* oidxf = perp + 1;                              // [NTOK]

    float*    wnorm  = (float*)d_ws;                      // [NUM_EMB]
    int*      idx    = (int*)(wnorm + NUM_EMB);           // [NTOK]
    unsigned* counts = (unsigned*)(idx + NTOK);           // [NUM_EMB]
    bf16_t*   WHg    = (bf16_t*)(counts + NUM_EMB);       // [NUM_EMB*DIM]
    bf16_t*   WLg    = WHg + (size_t)NUM_EMB * DIM;       // [NUM_EMB*DIM]

    size_t smem = (size_t)(2 * TOK_PER_WG * ROWPAD + 4 * 16 * ROWPAD) * sizeof(bf16_t); // 299,520 B
    (void)hipFuncSetAttribute((const void*)vq_argmin_kernel,
                              hipFuncAttributeMaxDynamicSharedMemorySize, (int)smem);

    zero_counts_kernel<<<4, 256, 0, stream>>>(counts);
    wsplit_kernel<<<(NUM_EMB * DIM / 4) / 256, 256, 0, stream>>>(W, WHg, WLg);
    wnorm_kernel<<<NUM_EMB / 8, 256, 0, stream>>>(W, wnorm);
    int nblk = (NTOK + TOK_PER_WG - 1) / TOK_PER_WG;      // 293 (tail block: 64 tokens)
    vq_argmin_kernel<<<nblk, 256, smem, stream>>>(X, WHg, WLg, wnorm, idx);
    vq_gather_kernel<<<NTOK / 8, 256, 0, stream>>>(X, W, idx, outq, loss, oidxf, counts);
    perplexity_kernel<<<1, 256, 0, stream>>>(counts, perp);
}
